// DynamicNetwork_24017457119877
// MI455X (gfx1250) — matrix operand order repaired and checked by reading, not compiled, not run on hardware
//
#include <hip/hip_runtime.h>
#include <hip/hip_bf16.h>

typedef _Float16 v16h __attribute__((ext_vector_type(16)));
typedef _Float16 v8h  __attribute__((ext_vector_type(8)));
typedef float    v8f  __attribute__((ext_vector_type(8)));
typedef float    v4f  __attribute__((ext_vector_type(4)));
typedef int      v4i_ __attribute__((ext_vector_type(4)));

typedef __attribute__((address_space(1))) v4i_* as1_v4i;
typedef __attribute__((address_space(3))) v4i_* as3_v4i;

#define NN      8192
#define A_FEAT  32
#define HH      64
#define LL      3
#define CUTOFF  0.5f
#define KSPLIT  8                  // K-range split across blockIdx.y

// LDS B-tile row stride in halves: 32 data halves padded to 40 (80 B) so the
// 16-lane ds_load_b128 pattern (stride 20 dwords) covers 16 distinct banks,
// and every 16B chunk stays 16B-aligned for b128 ds/async ops.
#define BROW    40
#define NROWS   (LL * HH)          // 192 (l,h) rows per K-tile

#if defined(__has_builtin)
#  if __has_builtin(__builtin_amdgcn_global_load_async_to_lds_b128)
#    define USE_ASYNC 1
#  endif
#endif
#ifndef USE_ASYNC
#  define USE_ASYNC 0
#endif

#if defined(__has_builtin) && __has_builtin(__builtin_amdgcn_s_wait_asynccnt)
#  define WAIT_ASYNC0() __builtin_amdgcn_s_wait_asynccnt(0)
#else
#  define WAIT_ASYNC0() asm volatile("s_wait_asynccnt 0x0" ::: "memory")
#endif

// ---------------------------------------------------------------------------
// ws layout:
//   floats [0, 192)           : colSum[l][h] accumulators (zeroed each launch)
//   byte offset 1024 onward   : tT as _Float16 [L][H][N]  (3 MB)
// ---------------------------------------------------------------------------

__global__ void zero_ws_kernel(float* __restrict__ p, int n) {
    int i = blockIdx.x * blockDim.x + threadIdx.x;
    if (i < n) p[i] = 0.0f;
}

// One 64-thread block per atom row i. Computes z_on (softplus), accumulates
// column sums of z_on, and writes t = z_on @ Wint + Bint transposed to f16
// tT[l][h][i] so the WMMA B operand is a contiguous K-run in memory.
__global__ __launch_bounds__(64) void onsite_kernel(
    const float* __restrict__ geom,
    const float* __restrict__ Won,
    const float* __restrict__ Bon,
    const float* __restrict__ Wint,
    const float* __restrict__ Bint,
    float* __restrict__ colSum,
    _Float16* __restrict__ tT)
{
    __shared__ float s_zon[HH];
    const int i = blockIdx.x;
    const int h = threadIdx.x;          // 0..63
    const float* zrow = geom + (size_t)i * A_FEAT;

    for (int l = 0; l < LL; ++l) {
        float acc = Bon[l * HH + h];
        #pragma unroll
        for (int a = 0; a < A_FEAT; ++a)
            acc += zrow[a] * Won[(l * A_FEAT + a) * HH + h];
        float sp = (acc > 20.0f) ? acc : __logf(1.0f + __expf(acc));
        __syncthreads();                // protect s_zon reuse across layers
        s_zon[h] = sp;
        atomicAdd(&colSum[l * HH + h], sp);
        __syncthreads();
        float t = Bint[l * HH + h];
        #pragma unroll
        for (int hh = 0; hh < HH; ++hh)
            t += s_zon[hh] * Wint[(l * HH + hh) * HH + h];
        tT[((size_t)(l * HH + h)) * NN + i] = (_Float16)t;
    }
}

// Stage one 3-layer B K-tile (192 rows x 32 halves) into LDS.
// 768 16B chunks, 128 threads -> 6 chunks/thread. Async path writes LDS
// directly via the ASYNCcnt engine; fallback is a plain reg round-trip.
__device__ __forceinline__ void stage_b_tile(const _Float16* __restrict__ tT,
                                             int jb, _Float16* dstBase, int tid)
{
    #pragma unroll
    for (int c = 0; c < 6; ++c) {
        const int chunk = tid + 128 * c;     // 0..767
        const int row   = chunk >> 2;        // (l*HH + h), 0..191
        const int q     = chunk & 3;         // 16B chunk within the 64B row
        const _Float16* gp = tT + (size_t)row * NN + jb + q * 8;
        _Float16*       lp = dstBase + row * BROW + q * 8;
#if USE_ASYNC
        _Float16* gpm = const_cast<_Float16*>(gp);
        __builtin_amdgcn_global_load_async_to_lds_b128(
            (as1_v4i)gpm, (as3_v4i)lp, 0, 0);
#else
        *(v8h*)lp = *(const v8h*)gp;
#endif
    }
}

// WMMA core: each wave owns a 16-row tile of S; K is split KSPLIT-ways across
// blockIdx.y and streamed in steps of 32 with a double-buffered LDS B tile
// shared by all 4 waves of the block.
__global__ __launch_bounds__(128) void interact_kernel(
    const float* __restrict__ dist,
    const _Float16* __restrict__ tT,
    const float* __restrict__ mu,
    const float* __restrict__ sigma,
    float* __restrict__ colSum)
{
    __shared__ _Float16 s_b[2][NROWS * BROW];

    const int tid     = threadIdx.x;
    const int lane    = tid & 31;
    const int wave    = tid >> 5;
    const int iBase   = (blockIdx.x * 4 + wave) * 16;
    const int r       = iBase + (lane & 15);   // this lane's A-matrix row (M)
    const int hiHalf  = lane >> 4;             // 0: lanes 0-15, 1: lanes 16-31
    const int k0a     = hiHalf * 16;           // A-operand K offset for this lane
    const int k0b     = hiHalf * 16;           // B-operand K offset for this lane
    const int jb0     = blockIdx.y * (NN / KSPLIT);
    const int jbEnd   = jb0 + (NN / KSPLIT);

    float inv_mu[LL], cexp[LL];
    #pragma unroll
    for (int l = 0; l < LL; ++l) {
        inv_mu[l] = 1.0f / mu[l];
        float s   = sigma[l];
        cexp[l]   = -1.0f / (2.0f * s * s);
    }

    v8f acc[LL][4];
    #pragma unroll
    for (int l = 0; l < LL; ++l)
        #pragma unroll
        for (int nt = 0; nt < 4; ++nt)
            acc[l][nt] = v8f{};

    const float* drow = dist + (size_t)r * NN;

    // Prologue: fill buffer 0.
    stage_b_tile(tT, jb0, &s_b[0][0], tid);
#if USE_ASYNC
    WAIT_ASYNC0();
#endif
    __syncthreads();

    int parity = 0;
    for (int jb = jb0; jb < jbEnd; jb += 32) {
        // Prefetch next B tile into the other buffer (uniform branch).
        if (jb + 32 < jbEnd)
            stage_b_tile(tT, jb + 32, &s_b[parity ^ 1][0], tid);

        const int j0 = jb + k0a;
        __builtin_prefetch(drow + j0 + 64, 0, 0);   // dist stream, 2 tiles ahead
        // 16 dist values per lane, in contiguous K order (B uses the same order):
        // halves 0..15 -> K = k0a+0..15
        v4f d0 = *(const v4f*)(drow + j0);
        v4f d1 = *(const v4f*)(drow + j0 + 4);
        v4f d2 = *(const v4f*)(drow + j0 + 8);
        v4f d3 = *(const v4f*)(drow + j0 + 12);
        float dd[16];
        #pragma unroll
        for (int q = 0; q < 4; ++q) {
            dd[q]      = d0[q];
            dd[4 + q]  = d1[q];
            dd[8 + q]  = d2[q];
            dd[12 + q] = d3[q];
        }

        // Phase 1: all 48 sensitivity values as f32 (independent TRANS ops).
        float sv[LL][16];
        #pragma unroll
        for (int m = 0; m < 16; ++m) {
            const int j    = j0 + m;
            const float d  = dd[m];
            const bool  ok = (d < CUTOFF) && (j != r);
            const float id = 1.0f / d;
            #pragma unroll
            for (int l = 0; l < LL; ++l) {
                float u = id - inv_mu[l];
                sv[l][m] = ok ? __expf(u * u * cexp[l]) : 0.0f;
            }
        }
        // Phase 2: pack to the three f16 A matrices.
        v16h Am[LL];
        #pragma unroll
        for (int l = 0; l < LL; ++l)
            #pragma unroll
            for (int m = 0; m < 16; ++m)
                Am[l][m] = (_Float16)sv[l][m];

        const _Float16* lbase = &s_b[parity][0] + (lane & 15) * BROW + k0b;
        #pragma unroll
        for (int l = 0; l < LL; ++l) {
            // Batch the 4 B operands of this layer so several ds_load_b128
            // stay in flight (partial dscnt waits) instead of a full drain
            // before every WMMA.
            v16h B0 = *(const v16h*)(lbase + (l * HH +  0) * BROW);
            v16h B1 = *(const v16h*)(lbase + (l * HH + 16) * BROW);
            v16h B2 = *(const v16h*)(lbase + (l * HH + 32) * BROW);
            v16h B3 = *(const v16h*)(lbase + (l * HH + 48) * BROW);
            acc[l][0] = __builtin_amdgcn_wmma_f32_16x16x32_f16(
                false, Am[l], false, B0, (short)0, acc[l][0], false, false);
            acc[l][1] = __builtin_amdgcn_wmma_f32_16x16x32_f16(
                false, Am[l], false, B1, (short)0, acc[l][1], false, false);
            acc[l][2] = __builtin_amdgcn_wmma_f32_16x16x32_f16(
                false, Am[l], false, B2, (short)0, acc[l][2], false, false);
            acc[l][3] = __builtin_amdgcn_wmma_f32_16x16x32_f16(
                false, Am[l], false, B3, (short)0, acc[l][3], false, false);
        }

#if USE_ASYNC
        WAIT_ASYNC0();      // next tile resident before anyone reads it
#endif
        __syncthreads();    // also fences reads of current buffer
        parity ^= 1;
    }

    // Only column sums of z_inter are needed: fold 16x16 C tiles.
    // C layout: lane holds column (lane&15); VGPR v = row v (lanes<16) / 8+v.
    #pragma unroll
    for (int l = 0; l < LL; ++l)
        #pragma unroll
        for (int nt = 0; nt < 4; ++nt) {
            v8f a = acc[l][nt];
            float s = a[0] + a[1] + a[2] + a[3] + a[4] + a[5] + a[6] + a[7];
            atomicAdd(&colSum[l * HH + nt * 16 + (lane & 15)], s);
        }
}

__global__ __launch_bounds__(64) void finalize_kernel(
    const float* __restrict__ colSum,
    const float* __restrict__ W_a,
    const float* __restrict__ B_n,
    const float* __restrict__ ppp,
    float* __restrict__ out)
{
    __shared__ float red[HH];
    const int h = threadIdx.x;           // 0..63
    float zc = 0.0f;
    #pragma unroll
    for (int l = 0; l < LL; ++l)
        zc += W_a[l * HH + h] * colSum[l * HH + h] + B_n[l];
    out[h] = ppp[h] + 0.01f * zc;
    red[h] = zc * zc;
    __syncthreads();
    if (h == 0) {
        float s = 0.0f;
        #pragma unroll
        for (int k = 0; k < HH; ++k) s += red[k];
        out[HH] = 0.01f * sqrtf(s);      // REG_STRENGTH * ||zc_sum||
    }
}

extern "C" void kernel_launch(void* const* d_in, const int* in_sizes, int n_in,
                              void* d_out, int out_size, void* d_ws, size_t ws_size,
                              hipStream_t stream) {
    const float* geom  = (const float*)d_in[0];
    const float* dist  = (const float*)d_in[1];
    const float* Won   = (const float*)d_in[2];
    const float* Bon   = (const float*)d_in[3];
    const float* Wint  = (const float*)d_in[4];
    const float* Bint  = (const float*)d_in[5];
    const float* mu    = (const float*)d_in[6];
    const float* sigma = (const float*)d_in[7];
    const float* W_a   = (const float*)d_in[8];
    const float* B_n   = (const float*)d_in[9];
    const float* ppp   = (const float*)d_in[10];
    float* out = (float*)d_out;

    float*     colSum = (float*)d_ws;
    _Float16*  tT     = (_Float16*)((char*)d_ws + 1024);

    zero_ws_kernel<<<1, 256, 0, stream>>>(colSum, LL * HH);
    onsite_kernel<<<NN, 64, 0, stream>>>(geom, Won, Bon, Wint, Bint, colSum, tT);
    // x: 128 row-blocks (4 waves x 16 rows); y: 8-way K split (dist read once).
    interact_kernel<<<dim3(128, KSPLIT), 128, 0, stream>>>(dist, tT, mu, sigma, colSum);
    finalize_kernel<<<1, 64, 0, stream>>>(colSum, W_a, B_n, ppp, out);
}
